// DenseLayer_65566970741071
// MI455X (gfx1250) — 8-bit matrix operand order repaired and checked by reading, compiled, not run on hardware
//
#include <hip/hip_runtime.h>
#include <stdint.h>
#include <stddef.h>

typedef int v8i __attribute__((ext_vector_type(8)));

#define M_DIM 8192
#define K_DIM 2048
#define N_DIM 2048
#define KT    (K_DIM / 64)
#define TILE_BYTES 1024

__global__ __launch_bounds__(256) void packA_kernel(const float* __restrict__ A,
                                                    uint32_t* __restrict__ Ap) {
    const uint32_t tile = blockIdx.x;
    const uint32_t kt   = tile % KT;
    const uint32_t mt   = tile / KT;
    const uint32_t w    = threadIdx.x;
    const uint32_t l    = w >> 3;
    const uint32_t v    = w & 7;
    const uint32_t m    = mt * 16 + (l & 15);
    const uint32_t k    = kt * 64 + (v >> 1) * 16 + (l >> 4) * 8 + (v & 1) * 4;

    const float* src = A + (size_t)m * K_DIM + k;
    uint32_t d = 0;
#pragma unroll
    for (int i = 0; i < 4; ++i) {
        const uint32_t s = (src[i] < 0.0f) ? 0xFFu : 0x01u;
        d |= s << (8 * i);
    }
    Ap[(size_t)tile * 256 + w] = d;
}

__global__ __launch_bounds__(256) void packB_kernel(const float* __restrict__ B,
                                                    uint32_t* __restrict__ Bp) {
    const uint32_t tile = blockIdx.x;
    const uint32_t kt   = tile % KT;
    const uint32_t nt   = tile / KT;
    const uint32_t w    = threadIdx.x;
    const uint32_t l    = w >> 3;
    const uint32_t v    = w & 7;
    const uint32_t k    = kt * 64 + (v >> 1) * 16 + (l >> 4) * 8 + (v & 1) * 4;
    const uint32_t n    = nt * 16 + (l & 15);

    const float* src = B + (size_t)k * N_DIM + n;
    uint32_t d = 0;
#pragma unroll
    for (int i = 0; i < 4; ++i) {
        const uint32_t s = (src[(size_t)i * N_DIM] < 0.0f) ? 0xFFu : 0x01u;
        d |= s << (8 * i);
    }
    Bp[(size_t)tile * 256 + w] = d;
}

__global__ __launch_bounds__(256) void bnn_gemm_kernel(
    const uint8_t* __restrict__ Ap, const uint8_t* __restrict__ Bp,
    const float* __restrict__ bias, float* __restrict__ out) {

    const int lane = threadIdx.x & 31;
    const int wave = threadIdx.x >> 5;
    const int wm   = wave >> 2;
    const int wn   = wave & 3;
    const int mt0  = blockIdx.y * 4 + wm * 2;
    const int nt0  = blockIdx.x * 8 + wn * 2;

    const uint8_t* a0p = Ap + (size_t)mt0 * KT * TILE_BYTES + lane * 32;
    const uint8_t* a1p = a0p + (size_t)KT * TILE_BYTES;
    const uint8_t* b0p = Bp + (size_t)nt0 * KT * TILE_BYTES + lane * 32;
    const uint8_t* b1p = b0p + (size_t)KT * TILE_BYTES;

    v8i c00 = {}, c01 = {}, c10 = {}, c11 = {};

#pragma unroll 2
    for (int kt = 0; kt < KT; ++kt) {
        const size_t off = (size_t)kt * TILE_BYTES;
        const v8i a0 = *(const v8i*)(a0p + off);
        const v8i a1 = *(const v8i*)(a1p + off);
        const v8i b0 = *(const v8i*)(b0p + off);
        const v8i b1 = *(const v8i*)(b1p + off);
        c00 = __builtin_amdgcn_wmma_i32_16x16x64_iu8(true, a0, true, b0, c00, false, false);
        c01 = __builtin_amdgcn_wmma_i32_16x16x64_iu8(true, a0, true, b1, c01, false, false);
        c10 = __builtin_amdgcn_wmma_i32_16x16x64_iu8(true, a1, true, b0, c10, false, false);
        c11 = __builtin_amdgcn_wmma_i32_16x16x64_iu8(true, a1, true, b1, c11, false, false);
    }

    const int   nl    = lane & 15;
    const int   mh    = (lane >> 4) * 8;
    const int   n0    = nt0 * 16 + nl;
    const int   n1    = (nt0 + 1) * 16 + nl;
    const float bias0 = bias[n0];
    const float bias1 = bias[n1];

#pragma unroll
    for (int r = 0; r < 8; ++r) {
        const size_t row0 = (size_t)(mt0 * 16 + mh + r) * N_DIM;
        const size_t row1 = (size_t)((mt0 + 1) * 16 + mh + r) * N_DIM;
        __builtin_nontemporal_store((float)c00[r] + bias0, out + row0 + n0);
        __builtin_nontemporal_store((float)c01[r] + bias1, out + row0 + n1);
        __builtin_nontemporal_store((float)c10[r] + bias0, out + row1 + n0);
        __builtin_nontemporal_store((float)c11[r] + bias1, out + row1 + n1);
    }
}

extern "C" void kernel_launch(void* const* d_in, const int* in_sizes, int n_in,
                              void* d_out, int out_size, void* d_ws, size_t ws_size,
                              hipStream_t stream) {
    (void)in_sizes; (void)n_in; (void)out_size; (void)ws_size;
    const float* A    = (const float*)d_in[0];
    const float* B    = (const float*)d_in[1];
    const float* bias = (const float*)d_in[2];
    float*       out  = (float*)d_out;

    uint32_t* Ap = (uint32_t*)d_ws;
    uint32_t* Bp = (uint32_t*)((uint8_t*)d_ws + (size_t)M_DIM * K_DIM);

    packA_kernel<<<(M_DIM / 16) * KT, 256, 0, stream>>>(A, Ap);
    packB_kernel<<<(N_DIM / 16) * KT, 256, 0, stream>>>(B, Bp);

    dim3 grid(N_DIM / 128, M_DIM / 64);
    bnn_gemm_kernel<<<grid, 256, 0, stream>>>((const uint8_t*)Ap, (const uint8_t*)Bp,
                                              bias, out);
}
